// RMDN_17789754540737
// MI455X (gfx1250) — matrix operand order repaired and checked by reading, not compiled, not run on hardware
//
#include <hip/hip_runtime.h>
#include <math.h>

// ---------------- problem constants ----------------
constexpr int Hc    = 1024;
constexpr int Ic    = 128;
constexpr int Oc    = 32;
constexpr int Fc    = 256;    // K*O
constexpr int Bc    = 128;
constexpr int Sc    = 512;
constexpr int FCIN  = 1568;   // H + 2F + O  (49 * 32)
constexpr int FCOUT = 2048;
constexpr int G3H   = 3072;

// d_out element offsets: (pis, mus, sgs, h_states) flattened
constexpr size_t OUT_PI = 0;
constexpr size_t OUT_MU = (size_t)Bc * Sc * 8;                 // 524288
constexpr size_t OUT_SG = OUT_MU + (size_t)Bc * Sc * Fc;       // 17301504
constexpr size_t OUT_H  = OUT_SG + (size_t)Bc * Sc * Fc;       // 34078720

typedef __bf16 bf16;
typedef __attribute__((ext_vector_type(16))) __bf16 v16bf;
typedef __attribute__((ext_vector_type(8)))  __bf16 v8bf;
typedef __attribute__((ext_vector_type(8)))  float  v8f;
typedef __attribute__((ext_vector_type(4)))  unsigned int u32x4;
typedef __attribute__((ext_vector_type(8)))  int i32x8;
typedef __attribute__((ext_vector_type(4)))  int i32x4;

#define DEV static __device__ __forceinline__

DEV bf16 f2bf(float f) { return (bf16)f; }   // native v_cvt path on gfx1250

DEV v8f wmma_bf16(v16bf a, v16bf b, v8f c) {
  // D = A(16x32) * B(32x16) + C, fp32 accum
  return __builtin_amdgcn_wmma_f32_16x16x32_bf16(
      /*neg_a=*/false, a, /*neg_b=*/false, b,
      /*c_mod=*/(short)0, c, /*reuse_a=*/false, /*reuse_b=*/false);
}

// ---------------- TDM: 1D/2D tile copy global -> LDS ----------------
// data_size = 8B elements. tile_d0 = elements per row, tile_d1 = rows (0/1 => 1D),
// row_stride = elements between rows in global memory.
DEV void tdm_load(const void* gptr, unsigned lds_off, unsigned tile_d0,
                  unsigned tile_d1, unsigned long row_stride) {
  unsigned long ga = (unsigned long)gptr;
  unsigned td1 = (tile_d1 > 1) ? tile_d1 : 1u;
  u32x4 g0;
  g0[0] = 1u;                                                // count=1, load, user
  g0[1] = lds_off;                                           // lds_addr (bytes)
  g0[2] = (unsigned)(ga & 0xFFFFFFFFu);                      // global_addr lo
  g0[3] = (unsigned)((ga >> 32) & 0x01FFFFFFu) | (2u << 30); // global hi | type=2
  i32x8 g1;
  g1[0] = (int)(3u << 16);                                   // data_size=3 (8B)
  g1[1] = (int)((tile_d0 & 0xFFFFu) << 16);                  // tensor_dim0 lo
  g1[2] = (int)((tile_d0 >> 16) | (td1 << 16));              // tensor_dim0 hi | tensor_dim1 lo
  g1[3] = (int)(tile_d0 << 16);                              // tensor_dim1 hi=0 | tile_dim0
  g1[4] = (int)((tile_d1 > 1) ? tile_d1 : 0u);               // tile_dim1 | tile_dim2=0
  g1[5] = (int)(row_stride & 0xFFFFFFFFu);                   // dim0_stride lo
  g1[6] = (int)((row_stride >> 32) & 0xFFFFu);               // dim0_stride hi
  g1[7] = 0;
  i32x4 z4 = {0, 0, 0, 0};
#if __clang_major__ >= 23
  i32x8 z8 = {0, 0, 0, 0, 0, 0, 0, 0};
  __builtin_amdgcn_tensor_load_to_lds(g0, g1, z4, z4, z8, 0);
#else
  __builtin_amdgcn_tensor_load_to_lds(g0, g1, z4, z4, 0);
#endif
}

// A fragment (16x32, MxK) from row-major fp32, converting to bf16.
DEV v16bf load_A_f32(const float* __restrict__ src, int ld, int m0, int k0, int lane) {
  const float* p = src + (size_t)(m0 + (lane & 15)) * ld + k0 + ((lane >> 4) << 3);
  float4 a = *(const float4*)(p + 0);
  float4 b = *(const float4*)(p + 4);
  float4 c = *(const float4*)(p + 16);
  float4 d = *(const float4*)(p + 20);
  v16bf r;
  r[0]=f2bf(a.x); r[1]=f2bf(a.y); r[2]=f2bf(a.z); r[3]=f2bf(a.w);
  r[4]=f2bf(b.x); r[5]=f2bf(b.y); r[6]=f2bf(b.z); r[7]=f2bf(b.w);
  r[8]=f2bf(c.x); r[9]=f2bf(c.y); r[10]=f2bf(c.z); r[11]=f2bf(c.w);
  r[12]=f2bf(d.x); r[13]=f2bf(d.y); r[14]=f2bf(d.z); r[15]=f2bf(d.w);
  return r;
}

// Same A fragment from row-major bf16 (works for global or LDS pointers).
DEV v16bf load_A_bf16(const bf16* __restrict__ src, int ld, int m0, int k0, int lane) {
  const bf16* p = src + (size_t)(m0 + (lane & 15)) * ld + k0 + ((lane >> 4) << 3);
  v8bf lo = *(const v8bf*)(p);
  v8bf hi = *(const v8bf*)(p + 16);
  v16bf r;
#pragma unroll
  for (int i = 0; i < 8; ++i) { r[i] = lo[i]; r[i + 8] = hi[i]; }
  return r;
}

// B fragment from pre-packed fragment-major weights (global or LDS).
DEV v16bf load_B(const bf16* __restrict__ base, int tileIdx, int lane) {
  return *(const v16bf*)(base + (((size_t)tileIdx << 5) + lane) * 16);
}

// ---------------- weight packing ----------------
__global__ void pack_b_kernel(const float* __restrict__ W, bf16* __restrict__ out,
                              int N, int Kdim, int Nreal) {
  int kT = Kdim >> 5;
  size_t total = (size_t)N * Kdim;
  size_t stride = (size_t)gridDim.x * blockDim.x;
  for (size_t idx = (size_t)blockIdx.x * blockDim.x + threadIdx.x; idx < total; idx += stride) {
    int    j    = (int)(idx & 15);
    size_t t    = idx >> 4;
    int    lane = (int)(t & 31);
    size_t tile = t >> 5;
    int    kt   = (int)(tile % kT);
    int    nt   = (int)(tile / kT);
    int    k    = kt * 32 + ((j & 8) << 1) + ((lane >> 4) << 3) + (j & 7);
    int    n    = nt * 16 + (lane & 15);
    float  v    = (n < Nreal) ? W[(size_t)n * Kdim + k] : 0.0f;
    out[idx] = f2bf(v);
  }
}

__global__ void init_kernel(unsigned* __restrict__ bar, bf16* __restrict__ musg) {
  int tid = blockIdx.x * blockDim.x + threadIdx.x;
  if (tid < 2) bar[tid] = 0u;
  int stride = gridDim.x * blockDim.x;
  for (int i = tid; i < Bc * 2 * Fc; i += stride) musg[i] = f2bf(0.0f);
}

__global__ void prep_y_kernel(const float* __restrict__ outputs, bf16* __restrict__ yB) {
  size_t total = (size_t)Sc * Bc * Oc;
  size_t stride = (size_t)gridDim.x * blockDim.x;
  for (size_t idx = (size_t)blockIdx.x * blockDim.x + threadIdx.x; idx < total; idx += stride) {
    int o  = (int)(idx & (Oc - 1));
    size_t t2 = idx >> 5;
    int b  = (int)(t2 & (Bc - 1));
    int s  = (int)(t2 >> 7);
    float v = (s == 0) ? 0.0f : outputs[((size_t)b * Sc + (s - 1)) * Oc + o];
    yB[idx] = f2bf(v);
  }
}

// ---------------- GEMM1 + GRU elementwise ----------------
__global__ __launch_bounds__(256)
void gru_kernel(const float* __restrict__ x, const float* __restrict__ b_ih,
                const float* __restrict__ b_hh, const bf16* __restrict__ WihP,
                bf16* __restrict__ hB, float* __restrict__ hOut) {
  int tid    = blockIdx.x * blockDim.x + threadIdx.x;
  int wave   = tid >> 5;
  int lane   = tid & 31;
  int nWaves = (gridDim.x * blockDim.x) >> 5;
  const int numTiles = (Bc * Sc / 16) * (Hc / 16);   // 4096 * 64

  for (int tile = wave; tile < numTiles; tile += nWaves) {
    int mt = tile >> 6;
    int nt = tile & 63;
    int m0 = mt * 16;
    v8f accR = {}, accZ = {}, accN = {};
#pragma unroll
    for (int kt = 0; kt < 4; ++kt) {
      v16bf a  = load_A_f32(x, Ic, m0, kt * 32, lane);
      v16bf br = load_B(WihP, (nt       ) * 4 + kt, lane);
      v16bf bz = load_B(WihP, (nt +  64 ) * 4 + kt, lane);
      v16bf bn = load_B(WihP, (nt + 128 ) * 4 + kt, lane);
      accR = wmma_bf16(a, br, accR);
      accZ = wmma_bf16(a, bz, accZ);
      accN = wmma_bf16(a, bn, accN);
    }
    int n   = lane & 15;
    int hi  = lane >> 4;
    int col = nt * 16 + n;
    float bir = b_ih[col], biz = b_ih[col + Hc], bin = b_ih[col + 2 * Hc];
    float bhr = b_hh[col], bhz = b_hh[col + Hc], bhn = b_hh[col + 2 * Hc];
#pragma unroll
    for (int r = 0; r < 8; ++r) {
      int m = m0 + r + hi * 8;                   // m = b*S + s
      float gr = accR[r] + bir + bhr;
      float gz = accZ[r] + biz + bhz;
      float rv = 1.0f / (1.0f + expf(-gr));
      float zv = 1.0f / (1.0f + expf(-gz));
      float nv = tanhf(accN[r] + bin + rv * bhn);
      float h  = (1.0f - zv) * nv;
      hOut[(size_t)m * Hc + col] = h;
      int b_ = m >> 9, s_ = m & (Sc - 1);
      hB[((size_t)s_ * Bc + b_) * Hc + col] = f2bf(h);
    }
  }
}

// ---------------- time-parallel base GEMM ----------------
__global__ __launch_bounds__(256)
void base_kernel(const bf16* __restrict__ hB, const bf16* __restrict__ yB,
                 const bf16* __restrict__ fcWP, const float* __restrict__ fc_b,
                 float* __restrict__ base) {
  int tid    = blockIdx.x * blockDim.x + threadIdx.x;
  int wave   = tid >> 5;
  int lane   = tid & 31;
  int nWaves = (gridDim.x * blockDim.x) >> 5;
  const int numTiles = (Sc * Bc / 16) * (FCOUT / 16);  // 4096 * 128
  const int kT = FCIN / 32;                            // 49

  for (int tile = wave; tile < numTiles; tile += nWaves) {
    int mt = tile >> 7;
    int nt = tile & 127;
    int m0 = mt * 16, n0 = nt * 16;
    v8f acc = {};
#pragma unroll 4
    for (int kt = 0; kt < 32; ++kt) {
      v16bf a = load_A_bf16(hB, Hc, m0, kt * 32, lane);
      v16bf b = load_B(fcWP, nt * kT + kt, lane);
      acc = wmma_bf16(a, b, acc);
    }
    {
      v16bf a = load_A_bf16(yB, Oc, m0, 0, lane);
      v16bf b = load_B(fcWP, nt * kT + 48, lane);
      acc = wmma_bf16(a, b, acc);
    }
    int n = lane & 15, hi = lane >> 4;
    float bias = fc_b[n0 + n];
#pragma unroll
    for (int r = 0; r < 8; ++r) {
      int m = m0 + r + hi * 8;
      base[(size_t)m * FCOUT + n0 + n] = acc[r] + bias;
    }
  }
}

// ---------------- grid barrier ----------------
DEV void grid_barrier(unsigned* bar, unsigned nblocks) {
  __threadfence();
  __syncthreads();
  if (threadIdx.x == 0) {
    volatile unsigned* gen = bar + 1;
    unsigned g = *gen;
    unsigned arrived = atomicAdd(bar, 1u);
    if (arrived == nblocks - 1u) {
      bar[0] = 0u;
      __threadfence();
      atomicAdd((unsigned*)(bar + 1), 1u);
    } else {
      while (*gen == g) { __builtin_amdgcn_s_sleep(2); }
    }
  }
  __syncthreads();
}

// ---------------- persistent scan kernel ----------------
// grid = 64 blocks x 256 threads (8 wave32). Block b owns N-tile columns {2b, 2b+1}.
// Per step: TDM-stage the 32KB fc_W panel (2D descriptor) and the 128KB mu/sg carry
// (1D descriptor) into LDS, s_wait_tensorcnt, then WMMA out of LDS.
__global__ __launch_bounds__(256)
void scan_kernel(const bf16* __restrict__ fcWP, const bf16* __restrict__ muWP,
                 const bf16* __restrict__ sgWP, const bf16* __restrict__ piWP,
                 const float* __restrict__ base, bf16* __restrict__ musg,
                 bf16* __restrict__ hsB,
                 const float* __restrict__ mu_b, const float* __restrict__ sg_b,
                 const float* __restrict__ pi_b,
                 float* __restrict__ outPi, float* __restrict__ outMu,
                 float* __restrict__ outSg, unsigned* __restrict__ bar) {
  extern __shared__ char smem[];
  bf16* ldsB = (bf16*)smem;                    // 2 nt * 16 tiles * 512 elem = 32KB
  bf16* ldsA = (bf16*)(smem + 32768);          // mu/sg carry: 128 x 512 bf16 = 128KB

  int tid    = threadIdx.x;
  int wid    = tid >> 5;                       // 0..7
  int lane   = tid & 31;
  int wave   = (blockIdx.x * blockDim.x + tid) >> 5;
  int nWaves = (gridDim.x * blockDim.x) >> 5;
  const int kTfc = FCIN / 32;                  // 49
  int n  = lane & 15;
  int hi = lane >> 4;
  int nt0 = blockIdx.x * 2;

  for (int t = 0; t < Sc; ++t) {
    const bf16* musgPrev = musg + (size_t)(t & 1) * (Bc * 2 * Fc);
    bf16*       musgNext = musg + (size_t)((t + 1) & 1) * (Bc * 2 * Fc);
    const float* baseT   = base + (size_t)t * Bc * FCOUT;

    // ---- TDM staging (issued once, by wave 0) ----
    if (tid < 32) {
      // fc_W panel: rows nt0..nt0+1, each row = 16 contiguous 1KB fragments,
      // global row stride = 49 fragments = 50176B = 6272 x 8B elements.
      tdm_load(fcWP + ((size_t)(nt0 * kTfc + 32) << 9),
               (unsigned)(size_t)ldsB, /*tile_d0=*/2048, /*tile_d1=*/2,
               /*row_stride=*/6272);
      // mu/sg carry: 128KB contiguous = 16384 x 8B elements (1D).
      tdm_load(musgPrev, (unsigned)(size_t)ldsA, /*tile_d0=*/16384, /*tile_d1=*/0,
               /*row_stride=*/16384);
      __builtin_amdgcn_s_wait_tensorcnt(0);
    }
    __syncthreads();

    // ---- phase A: hs = tanh(base + [mu_p|sg_p] @ fcW_ms^T), all from LDS ----
#pragma unroll
    for (int j = 0; j < 2; ++j) {
      int nt = nt0 + j;
      int m0 = wid * 16, n0 = nt * 16;
      v8f acc;
#pragma unroll
      for (int r = 0; r < 8; ++r)
        acc[r] = baseT[(size_t)(m0 + r + hi * 8) * FCOUT + n0 + n];
#pragma unroll 4
      for (int kt = 0; kt < 16; ++kt) {
        v16bf a = load_A_bf16(ldsA, 2 * Fc, m0, kt * 32, lane);
        v16bf b = load_B(ldsB, j * 16 + kt, lane);
        acc = wmma_bf16(a, b, acc);
      }
#pragma unroll
      for (int r = 0; r < 8; ++r)
        hsB[(size_t)(m0 + r + hi * 8) * FCOUT + n0 + n] = f2bf(tanhf(acc[r]));
    }
    grid_barrier(bar, gridDim.x);

    // ---- phase B: heads (mu: 128 tiles, sigma: 128, pi: 8) ----
    for (int tile = wave; tile < 264; tile += nWaves) {
      if (tile < 256) {
        bool isMu = tile < 128;
        int tt = isMu ? tile : tile - 128;
        int mt = tt >> 4;
        int nt = tt & 15;
        const bf16* Wp = isMu ? muWP : sgWP;
        v8f acc = {};
#pragma unroll 4
        for (int kt = 0; kt < 64; ++kt) {
          if (kt + 1 < 64)
            __builtin_prefetch(Wp + (((size_t)(nt * 64 + kt + 1) << 5) + lane) * 16, 0, 3);
          v16bf a = load_A_bf16(hsB, FCOUT, mt * 16, kt * 32, lane);
          v16bf b = load_B(Wp, nt * 64 + kt, lane);
          acc = wmma_bf16(a, b, acc);
        }
        int col = nt * 16 + n;
        float bias = isMu ? mu_b[col] : sg_b[col];
        float* dst = isMu ? outMu : outSg;
        int msOff  = isMu ? 0 : Fc;
#pragma unroll
        for (int r = 0; r < 8; ++r) {
          int bRow = mt * 16 + r + hi * 8;
          float v = acc[r] + bias;
          if (!isMu) {
            float sp = (v > 20.0f) ? v : log1pf(expf(v));
            v = fmaxf(sp, 0.001f);
          }
          dst[((size_t)bRow * Sc + t) * Fc + col] = v;
          musgNext[(size_t)bRow * (2 * Fc) + msOff + col] = f2bf(v);
        }
      } else {
        int mt = tile - 256;
        v8f acc = {};
#pragma unroll 4
        for (int kt = 0; kt < 64; ++kt) {
          v16bf a = load_A_bf16(hsB, FCOUT, mt * 16, kt * 32, lane);
          v16bf b = load_B(piWP, kt, lane);
          acc = wmma_bf16(a, b, acc);
        }
        float bias = (n < 8) ? pi_b[n] : 0.0f;
#pragma unroll
        for (int r = 0; r < 8; ++r) {
          float v  = acc[r] + bias;
          float vm = (n < 8) ? v : -1e30f;
          for (int off = 1; off < 8; off <<= 1)
            vm = fmaxf(vm, __shfl_xor(vm, off, 32));
          float e = (n < 8) ? expf(v - vm) : 0.0f;
          float sum = e;
          for (int off = 1; off < 8; off <<= 1)
            sum += __shfl_xor(sum, off, 32);
          if (n < 8) {
            int bRow = mt * 16 + r + hi * 8;
            outPi[((size_t)bRow * Sc + t) * 8 + n] = e / sum;
          }
        }
      }
    }
    grid_barrier(bar, gridDim.x);
  }
}

// ---------------- host ----------------
extern "C" void kernel_launch(void* const* d_in, const int* in_sizes, int n_in,
                              void* d_out, int out_size, void* d_ws, size_t ws_size,
                              hipStream_t stream) {
  const float* x       = (const float*)d_in[0];
  const float* outputs = (const float*)d_in[1];
  const float* W_ih    = (const float*)d_in[2];
  const float* b_ih    = (const float*)d_in[3];
  const float* b_hh    = (const float*)d_in[4];
  const float* fc_W    = (const float*)d_in[5];
  const float* fc_b    = (const float*)d_in[6];
  const float* pi_W    = (const float*)d_in[7];
  const float* pi_b    = (const float*)d_in[8];
  const float* mu_W    = (const float*)d_in[9];
  const float* mu_b    = (const float*)d_in[10];
  const float* sg_W    = (const float*)d_in[11];
  const float* sg_b    = (const float*)d_in[12];

  float* out   = (float*)d_out;
  float* outPi = out + OUT_PI;
  float* outMu = out + OUT_MU;
  float* outSg = out + OUT_SG;
  float* outH  = out + OUT_H;

  char* ws = (char*)d_ws;
  size_t off = 256;
  auto take = [&](size_t bytes) { size_t o = off; off += (bytes + 255) & ~(size_t)255; return o; };
  unsigned* bar  = (unsigned*)ws;                                     // 2 words
  bf16* WihP = (bf16*)(ws + take((size_t)G3H * Ic * 2));
  bf16* fcWP = (bf16*)(ws + take((size_t)FCOUT * FCIN * 2));
  bf16* muWP = (bf16*)(ws + take((size_t)Fc * FCOUT * 2));
  bf16* sgWP = (bf16*)(ws + take((size_t)Fc * FCOUT * 2));
  bf16* piWP = (bf16*)(ws + take((size_t)16 * FCOUT * 2));
  bf16* hB   = (bf16*)(ws + take((size_t)Sc * Bc * Hc * 2));
  bf16* yB   = (bf16*)(ws + take((size_t)Sc * Bc * Oc * 2));
  bf16* musg = (bf16*)(ws + take((size_t)2 * Bc * 2 * Fc * 2));
  bf16* hsB  = (bf16*)(ws + take((size_t)Bc * FCOUT * 2));
  float* baseP = (float*)(ws + take((size_t)Sc * Bc * FCOUT * 4));

  init_kernel<<<64, 256, 0, stream>>>(bar, musg);
  pack_b_kernel<<<1024, 256, 0, stream>>>(W_ih, WihP, G3H, Ic, G3H);
  pack_b_kernel<<<2048, 256, 0, stream>>>(fc_W, fcWP, FCOUT, FCIN, FCOUT);
  pack_b_kernel<<<1024, 256, 0, stream>>>(mu_W, muWP, Fc, FCOUT, Fc);
  pack_b_kernel<<<1024, 256, 0, stream>>>(sg_W, sgWP, Fc, FCOUT, Fc);
  pack_b_kernel<<<256, 256, 0, stream>>>(pi_W, piWP, 16, FCOUT, 8);
  prep_y_kernel<<<512, 256, 0, stream>>>(outputs, yB);

  gru_kernel<<<4096, 256, 0, stream>>>(x, b_ih, b_hh, WihP, hB, outH);
  base_kernel<<<4096, 256, 0, stream>>>(hB, yB, fcWP, fc_b, baseP);

  // 32KB B-panel + 128KB carry staged in LDS per block (320KB/WGP budget).
  scan_kernel<<<64, 256, 163840, stream>>>(fcWP, muWP, sgWP, piWP, baseP, musg, hsB,
                                           mu_b, sg_b, pi_b, outPi, outMu, outSg, bar);
}
